// LogDetter_82102594830418
// MI455X (gfx1250) — matrix operand order repaired and checked by reading, not compiled, not run on hardware
//
#include <hip/hip_runtime.h>
#include <hip/hip_bf16.h>
#include <math.h>

// ---------------------------------------------------------------------------
// logdet(x^T x) - D*log(N) for x[131072, 512] fp32.
//
// Stage 1 (FLOP-dominant, 68.7 GFLOP): G = x^T x via bf16 WMMA with hi/lo
//   split (G ~= H^T H + H^T L + L^T H) -> ~fp32 accuracy at bf16-WMMA rate.
//   Split-K partials into d_ws slabs (deterministic, no float atomics).
// Stage 2: deterministic slab reduction -> symmetric G (512x512) in d_ws.
// Stage 3: blocked Cholesky (NB=128, 4 steps: potrf/trsm/syrk kernels).
// Stage 4: logdet = 2*sum(log(diag(L))) - 512*log(131072) -> d_out[0].
//
// Workspace assumption: ws_size >= (SLABS+1)*512*512*4 bytes (~28.3 MB).
// ---------------------------------------------------------------------------

typedef __attribute__((ext_vector_type(16))) __bf16        v16bf;
typedef __attribute__((ext_vector_type(8)))  float         v8f;
typedef __attribute__((ext_vector_type(4)))  unsigned int  v4u;

#define N_ROWS  131072
#define D_COLS  512
#define SLABS   26
#define KCHUNKS (N_ROWS / 32)          // 4096 chunks of 32 rows
#define G_ELEMS (D_COLS * D_COLS)      // 262144
#define LDS_CSTRIDE 40                 // halfwords per staged column (32 + pad, keeps 16B align)

__device__ __forceinline__ unsigned short f32_to_bf16_rne(float f) {
    unsigned int u = __float_as_uint(f);
    unsigned int r = u + 0x7FFFu + ((u >> 16) & 1u);   // round-to-nearest-even
    return (unsigned short)(r >> 16);
}
__device__ __forceinline__ float bf16_to_f32(unsigned short h) {
    return __uint_as_float(((unsigned int)h) << 16);
}

union Frag { v16bf v; v4u q[2]; };

// ---------------------------------------------------------------------------
// Stage 1: partial Gram. grid = (SLABS, 10 lower-triangle 128x128 blocks).
// 256 threads = 8 waves; wave w computes the 16-row strip w of the block
// (8 accumulator tiles of 16x16 across the 128 block columns).
// ---------------------------------------------------------------------------
__global__ __launch_bounds__(256) void gram_partial(const float* __restrict__ x,
                                                    float* __restrict__ slabs) {
    __shared__ __align__(16) unsigned short ldsHi[256 * LDS_CSTRIDE]; // 20 KB
    __shared__ __align__(16) unsigned short ldsLo[256 * LDS_CSTRIDE]; // 20 KB

    const int s    = blockIdx.x;               // split-K slab id
    const int pair = blockIdx.y;               // lower-triangle block id (0..9)
    int bi = 0; while ((bi + 1) * (bi + 2) / 2 <= pair) ++bi;
    const int bj = pair - bi * (bi + 1) / 2;

    const int tid  = threadIdx.x;
    const int wave = tid >> 5;
    const int lane = tid & 31;
    const int l16  = lane & 15;
    const bool hih = (lane >= 16);

    v8f acc[8];
    #pragma unroll
    for (int t = 0; t < 8; ++t)
        #pragma unroll
        for (int e = 0; e < 8; ++e) acc[t][e] = 0.0f;

    for (int kc = s; kc < KCHUNKS; kc += SLABS) {
        const int n0 = kc * 32;
        __syncthreads();
        // Stage 32 rows x (block-i cols 0..127 | block-j cols 128..255) into
        // LDS column-major as bf16 hi/lo. Coalesced: thread t reads column
        // (t mod 256) of the fused range for each of the 32 rows.
        {
            const int c     = tid;                         // 0..255
            const int range = c >> 7;
            const int colg  = (range ? bj : bi) * 128 + (c & 127);
            #pragma unroll 4
            for (int k = 0; k < 32; ++k) {
                float v = x[(size_t)(n0 + k) * D_COLS + colg];
                unsigned short h = f32_to_bf16_rne(v);
                unsigned short l = f32_to_bf16_rne(v - bf16_to_f32(h));
                ldsHi[c * LDS_CSTRIDE + k] = h;
                ldsLo[c * LDS_CSTRIDE + k] = l;
            }
        }
        __syncthreads();

        // A fragment: 16-bit A 16x32 layout. lane<16: K{0..7,16..23};
        // lane>=16: K{8..15,24..31}; M = lane%16 -> x column wave*16+M.
        const int colA = wave * 16 + l16;
        const int aoff = colA * LDS_CSTRIDE + (hih ? 8 : 0);
        Frag aH, aL;
        aH.q[0] = *(const v4u*)&ldsHi[aoff];
        aH.q[1] = *(const v4u*)&ldsHi[aoff + 16];
        aL.q[0] = *(const v4u*)&ldsLo[aoff];
        aL.q[1] = *(const v4u*)&ldsLo[aoff + 16];

        #pragma unroll
        for (int tn = 0; tn < 8; ++tn) {
            // B fragment: 16-bit B 32x16 layout. lane<16: K{0..7,16..23},
            // lane>=16: K{8..15,24..31}; N = lane%16 -> x column 128 + tn*16 + N.
            const int colB = 128 + tn * 16 + l16;
            const int boff = colB * LDS_CSTRIDE + (hih ? 8 : 0);
            Frag bH, bL;
            bH.q[0] = *(const v4u*)&ldsHi[boff];
            bH.q[1] = *(const v4u*)&ldsHi[boff + 16];
            bL.q[0] = *(const v4u*)&ldsLo[boff];
            bL.q[1] = *(const v4u*)&ldsLo[boff + 16];

            acc[tn] = __builtin_amdgcn_wmma_f32_16x16x32_bf16(
                false, aH.v, false, bH.v, (short)0, acc[tn], false, false);
            acc[tn] = __builtin_amdgcn_wmma_f32_16x16x32_bf16(
                false, aH.v, false, bL.v, (short)0, acc[tn], false, false);
            acc[tn] = __builtin_amdgcn_wmma_f32_16x16x32_bf16(
                false, aL.v, false, bH.v, (short)0, acc[tn], false, false);
        }
    }

    // C/D layout: VGPR r holds M = r + (lane<16 ? 0 : 8), N = lane%16.
    float* slab = slabs + (size_t)s * G_ELEMS;
    #pragma unroll
    for (int tn = 0; tn < 8; ++tn)
        #pragma unroll
        for (int r = 0; r < 8; ++r) {
            int gr = bi * 128 + wave * 16 + r + (hih ? 8 : 0);
            int gc = bj * 128 + tn * 16 + l16;
            slab[(size_t)gr * D_COLS + gc] = acc[tn][r];
        }
}

// ---------------------------------------------------------------------------
// Stage 2: deterministic reduction over slabs; mirror to full symmetric G.
// ---------------------------------------------------------------------------
__global__ __launch_bounds__(256) void gram_reduce(const float* __restrict__ slabs,
                                                   float* __restrict__ G) {
    int idx = blockIdx.x * blockDim.x + threadIdx.x;
    if (idx >= G_ELEMS) return;
    int i = idx >> 9, j = idx & 511;
    int bi = i >> 7, bj = j >> 7;
    if (bi < bj) return;                 // only computed lower blocks
    if (bi == bj && i < j) return;       // diag block: lower half only (no races)
    float s = 0.0f;
    for (int k = 0; k < SLABS; ++k) s += slabs[(size_t)k * G_ELEMS + idx];
    G[idx] = s;
    if (i != j) G[(size_t)j * D_COLS + i] = s;
}

// ---------------------------------------------------------------------------
// Stage 3: blocked Cholesky, NB = 128, 4 panel steps.
// ---------------------------------------------------------------------------
__global__ __launch_bounds__(256) void potrf128(float* __restrict__ G, int p) {
    __shared__ float A[128][129];
    const int tid = threadIdx.x, base = p * 128;
    for (int idx = tid; idx < 128 * 128; idx += 256) {
        int r = idx >> 7, c = idx & 127;
        A[r][c] = G[(size_t)(base + r) * D_COLS + base + c];
    }
    __syncthreads();
    for (int k = 0; k < 128; ++k) {
        if (tid == 0) A[k][k] = sqrtf(A[k][k]);
        __syncthreads();
        float d = A[k][k];
        for (int r = k + 1 + tid; r < 128; r += 256) A[r][k] /= d;
        __syncthreads();
        int m = 127 - k;                            // symmetric square update
        for (int t = tid; t < m * m; t += 256) {
            int r = k + 1 + t / m, c = k + 1 + t % m;
            A[r][c] -= A[r][k] * A[c][k];
        }
        __syncthreads();
    }
    for (int idx = tid; idx < 128 * 128; idx += 256) {
        int r = idx >> 7, c = idx & 127;
        if (c <= r) G[(size_t)(base + r) * D_COLS + base + c] = A[r][c];
    }
}

__global__ __launch_bounds__(128) void trsm128(float* __restrict__ G, int p) {
    __shared__ float Lp[128][129];
    __shared__ float X[128][129];
    const int tid = threadIdx.x;
    const int i = p + 1 + blockIdx.x;
    const int pb = p * 128, ib = i * 128;
    for (int idx = tid; idx < 128 * 128; idx += 128) {
        int r = idx >> 7, c = idx & 127;
        Lp[r][c] = G[(size_t)(pb + r) * D_COLS + pb + c];
        X[r][c]  = G[(size_t)(ib + r) * D_COLS + pb + c];
    }
    __syncthreads();
    const int r = tid;                   // one row per thread, X = A * inv(Lpp^T)
    for (int c = 0; c < 128; ++c) {
        float sv = X[r][c];
        for (int k = 0; k < c; ++k) sv -= X[r][k] * Lp[c][k];
        X[r][c] = sv / Lp[c][c];
    }
    __syncthreads();
    for (int idx = tid; idx < 128 * 128; idx += 128) {
        int r2 = idx >> 7, c = idx & 127;
        G[(size_t)(ib + r2) * D_COLS + pb + c] = X[r2][c];
    }
}

__global__ __launch_bounds__(256) void syrk128(float* __restrict__ G, int p) {
    __shared__ float Li[128][129];
    __shared__ float Lj[128][129];
    int t = blockIdx.x;
    int ii = 0; while ((ii + 1) * (ii + 2) / 2 <= t) ++ii;
    int jj = t - ii * (ii + 1) / 2;
    const int i = p + 1 + ii, j = p + 1 + jj;
    const int tid = threadIdx.x;
    const int pb = p * 128, ib = i * 128, jb = j * 128;
    for (int idx = tid; idx < 128 * 128; idx += 256) {
        int r = idx >> 7, c = idx & 127;
        Li[r][c] = G[(size_t)(ib + r) * D_COLS + pb + c];
        Lj[r][c] = G[(size_t)(jb + r) * D_COLS + pb + c];
    }
    __syncthreads();
    for (int idx = tid; idx < 128 * 128; idx += 256) {
        int r = idx >> 7, c = idx & 127;
        float s = 0.0f;
        #pragma unroll 8
        for (int k = 0; k < 128; ++k) s += Li[r][k] * Lj[c][k];
        G[(size_t)(ib + r) * D_COLS + jb + c] -= s;
    }
}

// ---------------------------------------------------------------------------
// Stage 4: logdet = 2*sum(log(L_kk)) - D*log(N)
// ---------------------------------------------------------------------------
__global__ __launch_bounds__(512) void logdet_diag(const float* __restrict__ G,
                                                   float* __restrict__ out) {
    __shared__ float red[512];
    const int d = threadIdx.x;
    red[d] = 2.0f * logf(G[(size_t)d * D_COLS + d]);
    __syncthreads();
    for (int off = 256; off > 0; off >>= 1) {
        if (d < off) red[d] += red[d + off];
        __syncthreads();
    }
    if (d == 0) out[0] = red[0] - (float)D_COLS * logf((float)N_ROWS);
}

// ---------------------------------------------------------------------------
extern "C" void kernel_launch(void* const* d_in, const int* in_sizes, int n_in,
                              void* d_out, int out_size, void* d_ws, size_t ws_size,
                              hipStream_t stream) {
    (void)in_sizes; (void)n_in; (void)out_size; (void)ws_size;
    const float* x = (const float*)d_in[0];
    float* slabs = (float*)d_ws;                              // SLABS * 1 MB
    float* G     = slabs + (size_t)SLABS * G_ELEMS;           // 1 MB
    float* out   = (float*)d_out;

    dim3 ggrid(SLABS, 10);
    gram_partial<<<ggrid, 256, 0, stream>>>(x, slabs);
    gram_reduce<<<(G_ELEMS + 255) / 256, 256, 0, stream>>>(slabs, G);

    for (int p = 0; p < 4; ++p) {
        potrf128<<<1, 256, 0, stream>>>(G, p);
        const int rem = 3 - p;
        if (rem > 0) {
            trsm128<<<rem, 128, 0, stream>>>(G, p);
            syrk128<<<rem * (rem + 1) / 2, 256, 0, stream>>>(G, p);
        }
    }
    logdet_diag<<<1, 512, 0, stream>>>(G, out);
}
